// CANet_60035052863828
// MI455X (gfx1250) — matrix operand order repaired and checked by reading, not compiled, not run on hardware
//
#include <hip/hip_runtime.h>
#include <hip/hip_bf16.h>
#include <math.h>

typedef __attribute__((ext_vector_type(16))) __bf16 v16bf;
typedef __attribute__((ext_vector_type(8)))  __bf16 v8bf;
typedef __attribute__((ext_vector_type(8)))  float  v8f;
typedef __attribute__((ext_vector_type(4)))  float  v4f;

#define B_   4
#define C_   128
#define CI_  64
#define NQ   (96*96)    // 9216 query positions
#define NKV  (48*48)    // 2304 kv positions after 2x2 pool
#define QT   (NQ/16)    // 576
#define BN_EPS 1e-5f
#define NWA  (3*4*4*32) // packed A-tile chunks for projection weights

// D = A(16x32 bf16) * B(32x16 bf16) + C(16x16 f32)
static __device__ __forceinline__ v8f wmma_bf16(v16bf a, v16bf b, v8f c) {
  return __builtin_amdgcn_wmma_f32_16x16x32_bf16(false, a, false, b, (short)0, c, false, false);
}

// A-layout per lane (m = lane&15, h = lane>>4): elem i -> K = 16h+i.
// Caller passes p = row + 16h (+ kstep). Two aligned 16B loads.
static __device__ __forceinline__ v16bf load_a16(const __bf16* p) {
  v8bf lo = *(const v8bf*)p;
  v8bf hi = *(const v8bf*)(p + 8);
  return __builtin_shufflevector(lo, hi, 0,1,2,3,4,5,6,7,8,9,10,11,12,13,14,15);
}

static __device__ __forceinline__ unsigned pk_bf16(float a, float b) {
  union { __bf16 h[2]; unsigned u; } t;
  t.h[0] = (__bf16)a; t.h[1] = (__bf16)b;
  return t.u;
}

// ---------------------------------------------------------------------------
// Kernel W: one-shot weight prep.
//  - wA: theta/phi/g weights packed bf16 in per-lane A-tile order
//        wA[((j*4+t)*4+ks)*32 + lane][16]
//  - WwB: plain bf16 copy of W_w [C, CI] (B-layout is K-contiguous already)
// ---------------------------------------------------------------------------
__global__ void wprep_kernel(const float* __restrict__ tw, const float* __restrict__ pw,
                             const float* __restrict__ gw, const float* __restrict__ Ww,
                             __bf16* __restrict__ wA, __bf16* __restrict__ WwB)
{
  const int tid = blockIdx.x * blockDim.x + threadIdx.x;
  if (tid < NWA) {
    const int lane = tid & 31;
    const int ks   = (tid >> 5) & 3;
    const int t    = (tid >> 7) & 3;
    const int j    = tid >> 9;
    const float* w = (j == 0) ? tw : (j == 1) ? pw : gw;
    const int n  = lane & 15;
    const int kb = (lane >> 4) << 4;
    const float* src = w + (size_t)(t * 16 + n) * C_ + ks * 32 + kb;
    __bf16* dst = wA + (size_t)tid * 16;
    #pragma unroll
    for (int i = 0; i < 8; ++i) { dst[i] = (__bf16)src[i]; dst[8 + i] = (__bf16)src[8 + i]; }
  } else {
    const int k = tid - NWA;
    if (k < C_ * CI_) WwB[k] = (__bf16)Ww[k];
  }
}

// ---------------------------------------------------------------------------
// Kernel 0: transpose x [B,C,HW] f32 -> xT [B,HW,C] bf16 (LDS-tiled, coalesced)
// ---------------------------------------------------------------------------
__global__ void xpose_kernel(const float* __restrict__ x, __bf16* __restrict__ xT) {
  __shared__ __bf16 tile[32][33];
  const int b  = blockIdx.z;
  const int p0 = blockIdx.x * 32;
  const int c0 = blockIdx.y * 32;
  const int tx = threadIdx.x;
  #pragma unroll
  for (int i = threadIdx.y; i < 32; i += 8)
    tile[i][tx] = (__bf16)x[((size_t)b * C_ + c0 + i) * NQ + p0 + tx];
  __syncthreads();
  #pragma unroll
  for (int i = threadIdx.y; i < 32; i += 8)
    xT[((size_t)b * NQ + p0 + i) * C_ + c0 + tx] = tile[tx][i];
}

// ---------------------------------------------------------------------------
// Kernel 1: all three 1x1-conv projections as out^T = W * x^T.
// One wave: 16 pixels x (3 proj x 64 ci), K = 128.  A = packed wA (single
// 32B load per tile), B = xT rows (loaded once per k-step, reused by 12 WMMAs).
// D rows = ci, cols = pixel -> all stores contiguous in ci.
// ---------------------------------------------------------------------------
__global__ void __launch_bounds__(128)
proj_kernel(const __bf16* __restrict__ xT,
            const __bf16* __restrict__ wA,
            const float* __restrict__ tb, const float* __restrict__ pb,
            const float* __restrict__ gb,
            __bf16* __restrict__ theta_o,   // [B, NQ, CI] bf16
            float*  __restrict__ phi_o,     // [B, NQ, CI] f32 (pre-pool)
            float*  __restrict__ g_o)       // [B, NQ, CI] f32 (pre-pool)
{
  const int wid  = (blockIdx.x * blockDim.x + threadIdx.x) >> 5;
  const int lane = threadIdx.x & 31;
  if (wid >= B_ * QT) return;
  const int b  = wid / QT;
  const int p0 = (wid % QT) * 16;
  const int n  = lane & 15;
  const int h  = lane >> 4;
  const int kb = h << 3;

  const float* bs_[3] = { tb, pb, gb };

  v8f acc[3][4];
  #pragma unroll
  for (int j = 0; j < 3; ++j)
    #pragma unroll
    for (int t = 0; t < 4; ++t) {
      const float* bp = bs_[j] + t * 16 + kb;   // rows of C = ci
      #pragma unroll
      for (int r = 0; r < 8; ++r) acc[j][t][r] = bp[r];
    }

  const __bf16* xrow = xT + ((size_t)b * NQ + p0 + n) * C_;
  #pragma unroll
  for (int ks = 0; ks < 4; ++ks) {
    const v16bf bx = *(const v16bf*)(xrow + ks * 32 + 16 * h);   // B-tile, 32B aligned
    #pragma unroll
    for (int j = 0; j < 3; ++j)
      #pragma unroll
      for (int t = 0; t < 4; ++t) {
        const v16bf aw = *(const v16bf*)(wA + ((size_t)(((j * 4 + t) * 4 + ks) * 32) + lane) * 16);
        acc[j][t] = wmma_bf16(aw, bx, acc[j][t]);
      }
  }

  // D: row ci = t*16 + kb + r, col pixel = p0 + n  -> contiguous stores
  __bf16* to = theta_o + ((size_t)b * NQ + p0 + n) * CI_;
  float*  po = phi_o   + ((size_t)b * NQ + p0 + n) * CI_;
  float*  go = g_o     + ((size_t)b * NQ + p0 + n) * CI_;
  #pragma unroll
  for (int t = 0; t < 4; ++t) {
    v8bf pk;
    #pragma unroll
    for (int r = 0; r < 8; ++r) pk[r] = (__bf16)acc[0][t][r];
    *(v8bf*)(to + t * 16 + kb) = pk;
    #pragma unroll
    for (int r = 0; r < 8; ++r) po[t * 16 + kb + r] = acc[1][t][r];
    #pragma unroll
    for (int r = 0; r < 8; ++r) go[t * 16 + kb + r] = acc[2][t][r];
  }
}

// ---------------------------------------------------------------------------
// Kernel 2: 2x2 maxpool + bf16 pack.  phiT[b,kv,ci] (ci contig),
// gmat[b,ci,kv] (kv contig).  One thread = 8 ci values of one kv.
// ---------------------------------------------------------------------------
__global__ void pool_kernel(const float* __restrict__ phi_full,
                            const float* __restrict__ g_full,
                            __bf16* __restrict__ phiT,
                            __bf16* __restrict__ gmat)
{
  int idx = blockIdx.x * blockDim.x + threadIdx.x;
  if (idx >= B_ * NKV * 8) return;
  const int cg  = idx & 7;
  const int kv  = (idx >> 3) % NKV;
  const int b   = idx / (8 * NKV);
  const int ci0 = cg * 8;
  const int i = kv / 48, j = kv % 48;
  const int p00 = (2 * i) * 96 + 2 * j;
  const int pr[4] = { p00, p00 + 1, p00 + 96, p00 + 97 };
  const float* pf = phi_full + (size_t)b * NQ * CI_;
  const float* gf = g_full   + (size_t)b * NQ * CI_;

  float pm[8], gm[8];
  #pragma unroll
  for (int d = 0; d < 8; ++d) { pm[d] = -3.4e38f; gm[d] = -3.4e38f; }
  #pragma unroll
  for (int q = 0; q < 4; ++q) {
    const float* pr0 = pf + (size_t)pr[q] * CI_ + ci0;
    const float* gr0 = gf + (size_t)pr[q] * CI_ + ci0;
    #pragma unroll
    for (int d = 0; d < 8; ++d) { pm[d] = fmaxf(pm[d], pr0[d]); gm[d] = fmaxf(gm[d], gr0[d]); }
  }
  v8bf pk;
  #pragma unroll
  for (int d = 0; d < 8; ++d) pk[d] = (__bf16)pm[d];
  *(v8bf*)(phiT + ((size_t)b * NKV + kv) * CI_ + ci0) = pk;
  #pragma unroll
  for (int d = 0; d < 8; ++d)
    gmat[((size_t)b * CI_ + ci0 + d) * NKV + kv] = (__bf16)gm[d];
}

// ---------------------------------------------------------------------------
// Kernel 3: flash attention, transposed orientation.
// S^T = phi * theta^T : A = phiT rows (kv,ci), B = theta rows (q,ci).
// Softmax axis (kv) lands inside each lane -> in-register max/sum + one
// shfl_xor(16).  P^T (C-layout) converts to the PV B-layout with 8 packed
// half-wave exchanges.  y^T = G^T * P^T : A = gmat rows (ci,kv).
// One wave owns 32 query columns (2 n-tiles); 16 WMMAs per 32-kv chunk.
// ---------------------------------------------------------------------------
__global__ void __launch_bounds__(128)
attn_kernel(const __bf16* __restrict__ theta,
            const __bf16* __restrict__ phiT,
            const __bf16* __restrict__ gmat,
            __bf16* __restrict__ ymat)
{
  const int wid  = (blockIdx.x * blockDim.x + threadIdx.x) >> 5;
  const int lane = threadIdx.x & 31;
  if (wid >= B_ * (NQ / 32)) return;
  const int b  = wid / (NQ / 32);
  const int q0 = (wid % (NQ / 32)) * 32;
  const int n  = lane & 15;
  const int h  = lane >> 4;
  const int kb = h << 3;

  // theta B-tiles (columns = queries), loaded once
  v16bf bth[2][2];
  #pragma unroll
  for (int qt = 0; qt < 2; ++qt) {
    const __bf16* trow = theta + ((size_t)b * NQ + q0 + qt * 16 + n) * CI_;
    bth[qt][0] = *(const v16bf*)(trow + 16 * h);
    bth[qt][1] = *(const v16bf*)(trow + 32 + 16 * h);
  }

  float mrun[2] = { -1e30f, -1e30f };
  float lrun[2] = { 0.f, 0.f };
  v8f yacc[2][4];
  #pragma unroll
  for (int qt = 0; qt < 2; ++qt)
    #pragma unroll
    for (int t = 0; t < 4; ++t)
      #pragma unroll
      for (int r = 0; r < 8; ++r) yacc[qt][t][r] = 0.f;

  const __bf16* phib = phiT + (size_t)b * NKV * CI_;
  const __bf16* gbas = gmat + (size_t)b * CI_ * NKV;

  for (int kv0 = 0; kv0 < NKV; kv0 += 32) {
    // phi A-tiles: rows = kv
    v16bf aphi[2][2];
    #pragma unroll
    for (int sj = 0; sj < 2; ++sj) {
      const __bf16* prow = phib + (size_t)(kv0 + sj * 16 + n) * CI_;
      aphi[sj][0] = load_a16(prow + 16 * h);
      aphi[sj][1] = load_a16(prow + 32 + 16 * h);
    }

    v16bf pB[2];
    #pragma unroll
    for (int qt = 0; qt < 2; ++qt) {
      v8f s0, s1;
      #pragma unroll
      for (int r = 0; r < 8; ++r) { s0[r] = 0.f; s1[r] = 0.f; }
      s0 = wmma_bf16(aphi[0][0], bth[qt][0], s0);
      s0 = wmma_bf16(aphi[0][1], bth[qt][1], s0);
      s1 = wmma_bf16(aphi[1][0], bth[qt][0], s1);
      s1 = wmma_bf16(aphi[1][1], bth[qt][1], s1);

      // lane (n,h) holds S^T rows kv = {8h..8h+7} of each 16-row subtile,
      // all at column q = q0 + qt*16 + n.  Partner lane holds the rest.
      float pm = fmaxf(s0[0], s1[0]);
      #pragma unroll
      for (int r = 1; r < 8; ++r) pm = fmaxf(pm, fmaxf(s0[r], s1[r]));
      pm = fmaxf(pm, __shfl_xor(pm, 16, 32));
      const float nm = fmaxf(mrun[qt], pm);
      const float sc = __expf(mrun[qt] - nm);
      mrun[qt] = nm;

      float e0[8], e1[8], esum = 0.f;
      #pragma unroll
      for (int r = 0; r < 8; ++r) {
        e0[r] = __expf(s0[r] - nm);
        e1[r] = __expf(s1[r] - nm);
        esum += e0[r] + e1[r];
      }
      esum += __shfl_xor(esum, 16, 32);
      lrun[qt] = lrun[qt] * sc + esum;
      #pragma unroll
      for (int t = 0; t < 4; ++t)
        #pragma unroll
        for (int r = 0; r < 8; ++r) yacc[qt][t][r] *= sc;

      // assemble P^T in PV-B layout: elem i -> kv = 16h + i.
      // h=0 needs subtile0 rows 0..15 (own e0 + partner e0);
      // h=1 needs subtile1 rows 0..15 (partner e1 + own e1).
      unsigned pk0[4], pk1[4];
      #pragma unroll
      for (int j2 = 0; j2 < 4; ++j2) {
        pk0[j2] = pk_bf16(e0[2 * j2], e0[2 * j2 + 1]);
        pk1[j2] = pk_bf16(e1[2 * j2], e1[2 * j2 + 1]);
      }
      union { unsigned u[8]; v16bf v; } uu;
      #pragma unroll
      for (int j2 = 0; j2 < 4; ++j2) {
        unsigned x0 = (unsigned)__shfl_xor((int)pk0[j2], 16, 32);
        unsigned x1 = (unsigned)__shfl_xor((int)pk1[j2], 16, 32);
        uu.u[j2]     = h ? x1 : pk0[j2];
        uu.u[4 + j2] = h ? pk1[j2] : x0;
      }
      pB[qt] = uu.v;
    }

    // y^T += G^T * P^T : A = gmat rows (ci), reused for both q-subtiles
    #pragma unroll
    for (int t = 0; t < 4; ++t) {
      const __bf16* grow = gbas + (size_t)(t * 16 + n) * NKV + kv0;
      v16bf ag = load_a16(grow + 16 * h);
      yacc[0][t] = wmma_bf16(ag, pB[0], yacc[0][t]);
      yacc[1][t] = wmma_bf16(ag, pB[1], yacc[1][t]);
    }
  }

  // D rows = ci = t*16 + kb + r, cols = q -> contiguous bf16 stores
  #pragma unroll
  for (int qt = 0; qt < 2; ++qt) {
    __bf16* yo = ymat + ((size_t)b * NQ + q0 + qt * 16 + n) * CI_;
    const float rl = 1.f / lrun[qt];
    #pragma unroll
    for (int t = 0; t < 4; ++t) {
      v8bf pk;
      #pragma unroll
      for (int r = 0; r < 8; ++r) pk[r] = (__bf16)(yacc[qt][t][r] * rl);
      *(v8bf*)(yo + t * 16 + kb) = pk;
    }
  }
}

// ---------------------------------------------------------------------------
// Kernel 4: out = BN(conv1x1(y, W)) + x.  A = ymat rows (pixel), B = WwB rows
// (bf16, K-contig).  D rows = pixel -> contiguous f32 stores into [B,C,HW].
// ---------------------------------------------------------------------------
__global__ void __launch_bounds__(128)
out_kernel(const __bf16* __restrict__ ymat,
           const __bf16* __restrict__ WwB, const float* __restrict__ Wb,
           const float* __restrict__ gma, const float* __restrict__ bta,
           const float* __restrict__ mea, const float* __restrict__ var,
           const float* __restrict__ x,
           float* __restrict__ out)
{
  const int wid  = (blockIdx.x * blockDim.x + threadIdx.x) >> 5;
  const int lane = threadIdx.x & 31;
  if (wid >= B_ * QT * 2) return;
  const int b   = wid / (QT * 2);
  const int rem = wid % (QT * 2);
  const int o0  = (rem / QT) * 64;
  const int p0  = (rem % QT) * 16;
  const int n   = lane & 15;
  const int h   = lane >> 4;
  const int kb  = h << 3;

  const __bf16* yrow = ymat + ((size_t)b * NQ + p0 + n) * CI_;
  const v16bf a0 = load_a16(yrow + 16 * h);
  const v16bf a1 = load_a16(yrow + 32 + 16 * h);

  v8f acc[4];
  #pragma unroll
  for (int t = 0; t < 4; ++t) {
    const float bv = Wb[o0 + t * 16 + n];   // C cols = out channel
    #pragma unroll
    for (int r = 0; r < 8; ++r) acc[t][r] = bv;
  }
  #pragma unroll
  for (int t = 0; t < 4; ++t) {
    const __bf16* wr = WwB + (size_t)(o0 + t * 16 + n) * CI_;
    acc[t] = wmma_bf16(a0, *(const v16bf*)(wr + 16 * h), acc[t]);
    acc[t] = wmma_bf16(a1, *(const v16bf*)(wr + 32 + 16 * h), acc[t]);
  }

  const int rbase = p0 + kb;                 // D rows = pixel
  #pragma unroll
  for (int t = 0; t < 4; ++t) {
    const int o = o0 + t * 16 + n;
    const float inv = gma[o] * rsqrtf(var[o] + BN_EPS);
    const float add = bta[o] - mea[o] * inv;
    float*       op = out + ((size_t)b * C_ + o) * NQ + rbase;
    const float* xp = x   + ((size_t)b * C_ + o) * NQ + rbase;
    const v4f x0 = *(const v4f*)xp;
    const v4f x1 = *(const v4f*)(xp + 4);
    v4f r0, r1;
    #pragma unroll
    for (int r = 0; r < 4; ++r) {
      r0[r] = acc[t][r]     * inv + add + x0[r];
      r1[r] = acc[t][4 + r] * inv + add + x1[r];
    }
    *(v4f*)op = r0;
    *(v4f*)(op + 4) = r1;
  }
}

// ---------------------------------------------------------------------------
extern "C" void kernel_launch(void* const* d_in, const int* in_sizes, int n_in,
                              void* d_out, int out_size, void* d_ws, size_t ws_size,
                              hipStream_t stream) {
  const float* x   = (const float*)d_in[0];
  const float* gw  = (const float*)d_in[1];
  const float* gbv = (const float*)d_in[2];
  const float* tw  = (const float*)d_in[3];
  const float* tbv = (const float*)d_in[4];
  const float* pw  = (const float*)d_in[5];
  const float* pbv = (const float*)d_in[6];
  const float* Ww  = (const float*)d_in[7];
  const float* Wbv = (const float*)d_in[8];
  const float* gma = (const float*)d_in[9];
  const float* bta = (const float*)d_in[10];
  const float* mea = (const float*)d_in[11];
  const float* var = (const float*)d_in[12];
  float* out = (float*)d_out;

  char* ws = (char*)d_ws;
  size_t off = 0;
  auto alloc = [&](size_t bytes) -> char* {
    char* p = ws + off;
    off = (off + bytes + 255) & ~(size_t)255;
    return p;
  };
  __bf16* xT       = (__bf16*)alloc((size_t)B_ * NQ  * C_  * 2);
  __bf16* theta    = (__bf16*)alloc((size_t)B_ * NQ  * CI_ * 2);
  float*  phi_full = (float*) alloc((size_t)B_ * NQ  * CI_ * 4);
  float*  g_full   = (float*) alloc((size_t)B_ * NQ  * CI_ * 4);
  __bf16* phiT     = (__bf16*)alloc((size_t)B_ * NKV * CI_ * 2);
  __bf16* gmat     = (__bf16*)alloc((size_t)B_ * NKV * CI_ * 2);
  __bf16* ymat     = (__bf16*)alloc((size_t)B_ * NQ  * CI_ * 2);
  __bf16* wA       = (__bf16*)alloc((size_t)NWA * 16 * 2);
  __bf16* WwB      = (__bf16*)alloc((size_t)C_ * CI_ * 2);

  // W) one-shot weight pack (bf16, tile order)
  {
    const int threads = NWA + C_ * CI_;
    wprep_kernel<<<(threads + 255) / 256, 256, 0, stream>>>(tw, pw, gw, Ww, wA, WwB);
  }
  // 0) x -> xT bf16
  xpose_kernel<<<dim3(NQ / 32, C_ / 32, B_), dim3(32, 8), 0, stream>>>(x, xT);
  // 1) projections: 1 wave per 16-pixel tile (all three projections)
  {
    const int waves = B_ * QT;
    proj_kernel<<<(waves * 32 + 127) / 128, 128, 0, stream>>>(
        xT, wA, tbv, pbv, gbv, theta, phi_full, g_full);
  }
  // 2) 2x2 maxpool + layout
  {
    const int threads = B_ * NKV * 8;
    pool_kernel<<<(threads + 255) / 256, 256, 0, stream>>>(phi_full, g_full, phiT, gmat);
  }
  // 3) fused attention: 1 wave per 32-query tile
  {
    const int waves = B_ * (NQ / 32);
    attn_kernel<<<(waves * 32 + 127) / 128, 128, 0, stream>>>(theta, phiT, gmat, ymat);
  }
  // 4) output conv + BN + residual
  {
    const int waves = B_ * QT * 2;
    out_kernel<<<(waves * 32 + 127) / 128, 128, 0, stream>>>(
        ymat, WwB, Wbv, gma, bta, mea, var, x, out);
  }
}
